// DeformableCompositeTransformerDecoderLayer_52510270161225
// MI455X (gfx1250) — matrix operand order repaired and checked by reading, not compiled, not run on hardware
//
#include <hip/hip_runtime.h>
#include <math.h>

// ---------------------------------------------------------------------------
// Problem constants (match reference)
// ---------------------------------------------------------------------------
#define D_MODEL 256
#define D_FFN   1024
#define NH      8
#define HDIM    32
#define NLVL    4
#define NPNT    4
#define BSZ     4
#define NQ      100
#define NPTS    25
#define LQ      (NQ * NPTS)      // 2500
#define ROWS    (BSZ * LQ)       // 10000
#define LEN_IN  21760            // 128*128 + 64*64 + 32*32 + 16*16

typedef __attribute__((ext_vector_type(16))) _Float16 v16h;
typedef __attribute__((ext_vector_type(8)))  _Float16 v8h;
typedef __attribute__((ext_vector_type(8)))  float    v8f;

__constant__ int c_lvl_h[4] = {128, 64, 32, 16};
__constant__ int c_lvl_w[4] = {128, 64, 32, 16};
__constant__ int c_lvl_s[4] = {0, 16384, 20480, 21504};

// ---------------------------------------------------------------------------
// WMMA GEMM: C[M,N] = act(A[M,K] @ W[N,K]^T + bias[N])
// fp32 inputs, fp16 LDS tiles, fp32 accumulation via v_wmma_f32_16x16x32_f16.
// Block = 128 threads = 4 waves. Block tile 128x64, BK = 32.
// Wave w computes two 16-row strips (rows w*32..w*32+31) x 64 cols
//   -> 8 wmma per K step, each B fragment reused by 2 wmmas.
// Staging is branchless (A rows clamped to M-1; tile row r only feeds output
// row r, which is store-guarded) and fully vectorized:
//   global_load_b128 x2 -> cvt -> ds_store_b128 per 8 elements.
// LDS row stride = 40 halves (80 B) so every v8h access is 16B aligned.
// ---------------------------------------------------------------------------
__global__ __launch_bounds__(128)
void gemm_wmma_kernel(const float* __restrict__ A, const float* __restrict__ W,
                      const float* __restrict__ bias, float* __restrict__ C,
                      int M, int N, int K, int relu)
{
    __shared__ _Float16 As[128 * 40];
    __shared__ _Float16 Bs[64 * 40];

    const int tid  = threadIdx.x;
    const int lane = tid & 31;
    const int wv   = tid >> 5;
    const int row0 = blockIdx.y * 128;
    const int col0 = blockIdx.x * 64;

    const v8f vzero = {0.f, 0.f, 0.f, 0.f, 0.f, 0.f, 0.f, 0.f};
    v8f acc[2][4];
#pragma unroll
    for (int s = 0; s < 2; ++s)
#pragma unroll
        for (int j = 0; j < 4; ++j) acc[s][j] = vzero;

    // ISA 7.12.2 fragment addressing (wave32):
    // A 16x32 f16: lane<16 holds row=lane, halves = K{kbA..+7, kbA+16..+23}, kbA=0;
    //              lane>=16 same row (lane-16), kbA=8.
    const int aRowL = lane & 15;
    const int kbA   = (lane >> 4) * 8;
    // B 32x16 f16: lane<16 holds col=lane, K{0..7, 16..23}; lane>=16 K{8..15, 24..31}.
    const int bCol = lane & 15;
    const int kbB  = (lane >> 4) * 8;

    // Staging assignment: 4 threads per row, 8 columns each.
    const int sr = tid >> 2;          // 0..31
    const int sc = (tid & 3) * 8;     // 0, 8, 16, 24

    for (int k0 = 0; k0 < K; k0 += 32) {
        // ---- stage A (128 rows) ----
#pragma unroll
        for (int rr = 0; rr < 4; ++rr) {
            int r = sr + rr * 32;
            int gr = row0 + r; if (gr > M - 1) gr = M - 1;   // clamp: branchless
            const float4* ap = (const float4*)&A[(long long)gr * K + k0 + sc];
            float4 f0 = ap[0], f1 = ap[1];
            v8h hv;
            hv[0] = (_Float16)f0.x; hv[1] = (_Float16)f0.y;
            hv[2] = (_Float16)f0.z; hv[3] = (_Float16)f0.w;
            hv[4] = (_Float16)f1.x; hv[5] = (_Float16)f1.y;
            hv[6] = (_Float16)f1.z; hv[7] = (_Float16)f1.w;
            *(v8h*)&As[r * 40 + sc] = hv;
        }
        // ---- stage B / W (64 rows, always in range: N % 64 == 0) ----
#pragma unroll
        for (int rr = 0; rr < 2; ++rr) {
            int r = sr + rr * 32;
            const float4* wp = (const float4*)&W[(long long)(col0 + r) * K + k0 + sc];
            float4 f0 = wp[0], f1 = wp[1];
            v8h hv;
            hv[0] = (_Float16)f0.x; hv[1] = (_Float16)f0.y;
            hv[2] = (_Float16)f0.z; hv[3] = (_Float16)f0.w;
            hv[4] = (_Float16)f1.x; hv[5] = (_Float16)f1.y;
            hv[6] = (_Float16)f1.z; hv[7] = (_Float16)f1.w;
            *(v8h*)&Bs[r * 40 + sc] = hv;
        }
        if (k0 + 32 < K) {
            int gr = row0 + sr * 4; if (gr > M - 1) gr = M - 1;
            __builtin_prefetch(&A[(long long)gr * K + k0 + 32], 0, 1);
        }
        __syncthreads();

        v16h af[2];
#pragma unroll
        for (int s = 0; s < 2; ++s) {
            const int ar = wv * 32 + s * 16 + aRowL;
            const v8h lo = *(const v8h*)&As[ar * 40 + kbA];
            const v8h hi = *(const v8h*)&As[ar * 40 + kbA + 16];
#pragma unroll
            for (int t = 0; t < 8; ++t) { af[s][t] = lo[t]; af[s][t + 8] = hi[t]; }
        }
#pragma unroll
        for (int j = 0; j < 4; ++j) {
            const _Float16* bp = &Bs[(j * 16 + bCol) * 40 + kbB];
            const v8h lo = *(const v8h*)bp;
            const v8h hi = *(const v8h*)(bp + 16);
            v16h bf;
#pragma unroll
            for (int t = 0; t < 8; ++t) { bf[t] = lo[t]; bf[t + 8] = hi[t]; }
            acc[0][j] = __builtin_amdgcn_wmma_f32_16x16x32_f16(
                false, af[0], false, bf, (short)0, acc[0][j], false, false);
            acc[1][j] = __builtin_amdgcn_wmma_f32_16x16x32_f16(
                false, af[1], false, bf, (short)0, acc[1][j], false, false);
        }
        __syncthreads();
    }

    // C/D layout: VGPR v: lanes 0-15 -> M=v, N=lane; lanes 16-31 -> M=v+8, N=lane-16.
    const int cbase = col0 + (lane & 15);
#pragma unroll
    for (int s = 0; s < 2; ++s) {
        const int rbase = row0 + wv * 32 + s * 16 + ((lane < 16) ? 0 : 8);
#pragma unroll
        for (int j = 0; j < 4; ++j) {
            int col = cbase + j * 16;
            float bv = bias ? bias[col] : 0.f;
#pragma unroll
            for (int v = 0; v < 8; ++v) {
                int row = rbase + v;
                if (row < M) {
                    float x = acc[s][j][v] + bv;
                    if (relu) x = fmaxf(x, 0.f);
                    C[(long long)row * N + col] = x;
                }
            }
        }
    }
}

// ---------------------------------------------------------------------------
// Intra self-attention: 400 sequences of 25 tokens, 8 heads, HD=32.
// One block per (seq, head). qk: [ROWS,512] (q|k), v: [ROWS,256].
// ---------------------------------------------------------------------------
__global__ __launch_bounds__(128)
void attn_intra_kernel(const float* __restrict__ qk, const float* __restrict__ vb,
                       float* __restrict__ out)
{
    __shared__ float sq[25 * 32], sk[25 * 32], sv[25 * 32], sc[25 * 25];
    const int s = blockIdx.x >> 3;
    const int h = blockIdx.x & 7;
    const int tid = threadIdx.x;
    const int rb = s * 25;

    for (int i = tid; i < 25 * 32; i += 128) {
        int r = i >> 5, d = i & 31;
        sq[i] = qk[(long long)(rb + r) * 512 + h * 32 + d];
        sk[i] = qk[(long long)(rb + r) * 512 + 256 + h * 32 + d];
        sv[i] = vb[(long long)(rb + r) * 256 + h * 32 + d];
    }
    __syncthreads();

    const float scale = 0.17677669529663688f;  // 1/sqrt(32)
    for (int e = tid; e < 625; e += 128) {
        int i = e / 25, j = e % 25;
        float a = 0.f;
#pragma unroll
        for (int d = 0; d < 32; ++d) a += sq[i * 32 + d] * sk[j * 32 + d];
        sc[e] = a * scale;
    }
    __syncthreads();

    if (tid < 25) {
        float m = -1e30f;
        for (int j = 0; j < 25; ++j) m = fmaxf(m, sc[tid * 25 + j]);
        float s0 = 0.f;
        for (int j = 0; j < 25; ++j) { float e = __expf(sc[tid * 25 + j] - m); sc[tid * 25 + j] = e; s0 += e; }
        float inv = 1.f / s0;
        for (int j = 0; j < 25; ++j) sc[tid * 25 + j] *= inv;
    }
    __syncthreads();

    for (int e = tid; e < 800; e += 128) {
        int i = e >> 5, d = e & 31;
        float a = 0.f;
        for (int j = 0; j < 25; ++j) a += sc[i * 25 + j] * sv[j * 32 + d];
        out[(long long)(rb + i) * 256 + h * 32 + d] = a;
    }
}

// ---------------------------------------------------------------------------
// Inter self-attention: 100 sequences of 100 tokens, 8 heads.
// qkv: [ROWS,768] (q|k|v). One block per (seq, head); 32-row query chunks.
// ---------------------------------------------------------------------------
__global__ __launch_bounds__(256)
void attn_inter_kernel(const float* __restrict__ qkv, float* __restrict__ out)
{
    __shared__ float sk[100 * 32], sv[100 * 32], sq[32 * 32], sc[32 * 100];
    const int s = blockIdx.x >> 3;
    const int h = blockIdx.x & 7;
    const int tid = threadIdx.x;
    const int rb = s * 100;

    for (int i = tid; i < 100 * 32; i += 256) {
        int r = i >> 5, d = i & 31;
        sk[i] = qkv[(long long)(rb + r) * 768 + 256 + h * 32 + d];
        sv[i] = qkv[(long long)(rb + r) * 768 + 512 + h * 32 + d];
    }

    const float scale = 0.17677669529663688f;
    for (int qb = 0; qb < 4; ++qb) {
        __syncthreads();
        for (int i = tid; i < 32 * 32; i += 256) {
            int r = qb * 32 + (i >> 5);
            sq[i] = (r < 100) ? qkv[(long long)(rb + r) * 768 + h * 32 + (i & 31)] : 0.f;
        }
        __syncthreads();
        for (int e = tid; e < 3200; e += 256) {
            int i = e / 100, j = e % 100;
            float a = 0.f;
#pragma unroll
            for (int d = 0; d < 32; ++d) a += sq[i * 32 + d] * sk[j * 32 + d];
            sc[e] = a * scale;
        }
        __syncthreads();
        if (tid < 32 && qb * 32 + tid < 100) {
            float m = -1e30f;
            for (int j = 0; j < 100; ++j) m = fmaxf(m, sc[tid * 100 + j]);
            float s0 = 0.f;
            for (int j = 0; j < 100; ++j) { float e = __expf(sc[tid * 100 + j] - m); sc[tid * 100 + j] = e; s0 += e; }
            float inv = 1.f / s0;
            for (int j = 0; j < 100; ++j) sc[tid * 100 + j] *= inv;
        }
        __syncthreads();
        for (int e = tid; e < 32 * 32; e += 256) {
            int i = e >> 5, d = e & 31;
            int r = qb * 32 + i;
            if (r < 100) {
                float a = 0.f;
                for (int j = 0; j < 100; ++j) a += sc[i * 100 + j] * sv[j * 32 + d];
                out[(long long)(rb + r) * 256 + h * 32 + d] = a;
            }
        }
    }
}

// ---------------------------------------------------------------------------
// out = LayerNorm(x + res) * g + b, one wave32 per 256-wide row.
// ---------------------------------------------------------------------------
__global__ __launch_bounds__(256)
void residual_ln_kernel(const float* __restrict__ x, const float* __restrict__ res,
                        const float* __restrict__ g, const float* __restrict__ b,
                        float* __restrict__ out, int rows)
{
    const int wid = (int)((blockIdx.x * blockDim.x + threadIdx.x) >> 5);
    const int lane = threadIdx.x & 31;
    if (wid >= rows) return;

    float v[8];
    float sum = 0.f;
#pragma unroll
    for (int t = 0; t < 8; ++t) {
        int c = lane + t * 32;
        v[t] = x[(long long)wid * 256 + c] + res[(long long)wid * 256 + c];
        sum += v[t];
    }
#pragma unroll
    for (int o = 16; o > 0; o >>= 1) sum += __shfl_xor(sum, o, 32);
    float mean = sum * (1.f / 256.f);
    float var = 0.f;
#pragma unroll
    for (int t = 0; t < 8; ++t) { float d = v[t] - mean; var += d * d; }
#pragma unroll
    for (int o = 16; o > 0; o >>= 1) var += __shfl_xor(var, o, 32);
    var *= (1.f / 256.f);
    float inv = rsqrtf(var + 1e-5f);
#pragma unroll
    for (int t = 0; t < 8; ++t) {
        int c = lane + t * 32;
        out[(long long)wid * 256 + c] = (v[t] - mean) * inv * g[c] + b[c];
    }
}

__global__ void add_kernel(const float* __restrict__ a, const float* __restrict__ b,
                           float* __restrict__ c, int n)
{
    int i = blockIdx.x * blockDim.x + threadIdx.x;
    if (i < n) c[i] = a[i] + b[i];
}

// Permute [BSZ,NQ,NPTS,D] <-> [BSZ,NPTS,NQ,D]
__global__ void transpose_qp_kernel(const float* __restrict__ in, float* __restrict__ out, int fwd)
{
    long long i = (long long)blockIdx.x * blockDim.x + threadIdx.x;
    if (i >= (long long)ROWS * 256) return;
    int c = (int)(i & 255);
    long long r = i >> 8;
    int b = (int)(r / 2500);
    int rem = (int)(r % 2500);
    if (fwd) {  // out index (b,p,q), source (b,q,p)
        int p = rem / 100, q = rem % 100;
        out[i] = in[(((long long)b * 100 + q) * 25 + p) * 256 + c];
    } else {    // out index (b,q,p), source (b,p,q)
        int q = rem / 25, p = rem % 25;
        out[i] = in[(((long long)b * 25 + p) * 100 + q) * 256 + c];
    }
}

// Softmax over 16 (NLVL*NPNT) attention weights per (row, head); in-place.
__global__ void aw_softmax_kernel(float* __restrict__ aw)
{
    int g = blockIdx.x * blockDim.x + threadIdx.x;
    if (g >= ROWS * NH) return;
    float* p = aw + (long long)(g >> 3) * 128 + (g & 7) * 16;
    float m = -1e30f;
#pragma unroll
    for (int e = 0; e < 16; ++e) m = fmaxf(m, p[e]);
    float s = 0.f;
#pragma unroll
    for (int e = 0; e < 16; ++e) { float x = __expf(p[e] - m); p[e] = x; s += x; }
    float inv = 1.f / s;
#pragma unroll
    for (int e = 0; e < 16; ++e) p[e] *= inv;
}

// ---------------------------------------------------------------------------
// Multi-scale deformable sampling. One wave per (b, lq, head); lane = channel.
// value: [BSZ, LEN_IN, 256] (col = h*32+d); off: [ROWS,256]; aw: [ROWS,128].
// ---------------------------------------------------------------------------
__global__ __launch_bounds__(256)
void deform_sample_kernel(const float* __restrict__ value, const float* __restrict__ off,
                          const float* __restrict__ aw, const float* __restrict__ refp,
                          float* __restrict__ out)
{
    const int g = (int)((blockIdx.x * blockDim.x + threadIdx.x) >> 5);
    const int lane = threadIdx.x & 31;
    if (g >= ROWS * NH) return;
    const int h = g & 7;
    const long long row = g >> 3;         // b*2500 + lq
    const int b = (int)(row / 2500);
    const int lq = (int)(row % 2500);
    const int q = lq / 25;

    const float* orow = off + row * 256;
    const float* arow = aw + row * 128 + h * 16;
    float acc = 0.f;

    for (int l = 0; l < NLVL; ++l) {
        const int ww = c_lvl_w[l], hh = c_lvl_h[l];
        const float wl = (float)ww, hl = (float)hh;
        const float rx = refp[(((long long)b * 100 + q) * 4 + l) * 2 + 0];
        const float ry = refp[(((long long)b * 100 + q) * 4 + l) * 2 + 1];
        const float* vbase = value + ((long long)b * LEN_IN + c_lvl_s[l]) * 256 + h * 32 + lane;
#pragma unroll
        for (int p = 0; p < NPNT; ++p) {
            const int oidx = ((h * 4 + l) * 4 + p) * 2;
            const float x = (rx + orow[oidx + 0] / wl) * wl - 0.5f;
            const float y = (ry + orow[oidx + 1] / hl) * hl - 0.5f;
            const float a = arow[l * 4 + p];
            const float x0f = floorf(x), y0f = floorf(y);
            const int x0 = (int)x0f, y0 = (int)y0f;
            const float wx = x - x0f, wy = y - y0f;
            float s = 0.f;
#pragma unroll
            for (int dy = 0; dy < 2; ++dy) {
#pragma unroll
                for (int dx = 0; dx < 2; ++dx) {
                    int xi = x0 + dx, yi = y0 + dy;
                    if (xi >= 0 && xi < ww && yi >= 0 && yi < hh) {
                        float wgt = (dx ? wx : 1.f - wx) * (dy ? wy : 1.f - wy);
                        s += wgt * vbase[(long long)(yi * ww + xi) * 256];
                    }
                }
            }
            acc += a * s;
        }
    }
    out[row * 256 + h * 32 + lane] = acc;
}

// ---------------------------------------------------------------------------
// Host orchestration
// ---------------------------------------------------------------------------
extern "C" void kernel_launch(void* const* d_in, const int* in_sizes, int n_in,
                              void* d_out, int out_size, void* d_ws, size_t ws_size,
                              hipStream_t stream)
{
    (void)in_sizes; (void)n_in; (void)out_size; (void)ws_size;

    const float* tgt    = (const float*)d_in[0];
    const float* qpos   = (const float*)d_in[1];
    const float* refp   = (const float*)d_in[2];
    const float* src    = (const float*)d_in[3];
    const float* w_in_a = (const float*)d_in[4];
    const float* b_in_a = (const float*)d_in[5];
    const float* w_out_a= (const float*)d_in[6];
    const float* b_out_a= (const float*)d_in[7];
    const float* g_ln_a = (const float*)d_in[8];
    const float* b_ln_a = (const float*)d_in[9];
    const float* w_in_i = (const float*)d_in[10];
    const float* b_in_i = (const float*)d_in[11];
    const float* w_out_i= (const float*)d_in[12];
    const float* b_out_i= (const float*)d_in[13];
    const float* g_ln_i = (const float*)d_in[14];
    const float* b_ln_i = (const float*)d_in[15];
    const float* off_w  = (const float*)d_in[16];
    const float* off_b  = (const float*)d_in[17];
    const float* aw_w   = (const float*)d_in[18];
    const float* aw_b   = (const float*)d_in[19];
    const float* val_w  = (const float*)d_in[20];
    const float* val_b  = (const float*)d_in[21];
    const float* co_w   = (const float*)d_in[22];
    const float* co_b   = (const float*)d_in[23];
    const float* g_ln_c = (const float*)d_in[24];
    const float* b_ln_c = (const float*)d_in[25];
    const float* lin1_w = (const float*)d_in[26];
    const float* lin1_b = (const float*)d_in[27];
    const float* lin2_w = (const float*)d_in[28];
    const float* lin2_b = (const float*)d_in[29];
    const float* g_ln3  = (const float*)d_in[30];
    const float* b_ln3  = (const float*)d_in[31];

    float* ws = (float*)d_ws;
    size_t o = 0;
    float* VAL = ws + o; o += (size_t)BSZ * LEN_IN * 256;  // value projection
    float* SA  = ws + o; o += (size_t)ROWS * 256;          // q-inputs (tgt+pos / qc)
    float* SB  = ws + o; o += (size_t)ROWS * 512;          // intra q|k
    float* SC  = ws + o; o += (size_t)ROWS * 256;          // intra v
    float* SD  = ws + o; o += (size_t)ROWS * 256;          // attn out / sampled
    float* SE  = ws + o; o += (size_t)ROWS * 256;          // GEMM t2 (pre-LN)
    float* SF  = ws + o; o += (size_t)ROWS * 256;          // LN results
    float* SG  = ws + o; o += (size_t)ROWS * 256;          // ti / ti_back (residual)
    float* SH  = ws + o; o += (size_t)ROWS * 1024;         // inter qkv / ffn hidden
    float* SI  = ws + o; o += (size_t)ROWS * 128;          // attention weights
    float* SJ  = ws + o; o += (size_t)ROWS * 256;          // sampling offsets
    float* SK  = ws + o; o += (size_t)ROWS * 256;          // tgt after cross LN

    const int NEL = ROWS * 256;          // 2,560,000
    const int EW_BLOCKS = (NEL + 255) / 256;
    const int LN_BLOCKS = (ROWS + 7) / 8;            // 8 waves/block
    const int SAMP_BLOCKS = (ROWS * NH + 7) / 8;     // 8 waves/block

    auto gemm = [&](const float* A, const float* W, const float* bias, float* C,
                    int M, int N, int K, int relu) {
        dim3 grid(N / 64, (M + 127) / 128);
        gemm_wmma_kernel<<<grid, 128, 0, stream>>>(A, W, bias, C, M, N, K, relu);
    };

    // ---- value projection (biggest GEMM, M = 87040) ----
    gemm(src, val_w, val_b, VAL, BSZ * LEN_IN, 256, 256, 0);

    // ---- intra-instance self-attention over control points ----
    add_kernel<<<EW_BLOCKS, 256, 0, stream>>>(tgt, qpos, SA, NEL);
    gemm(SA, w_in_a, b_in_a, SB, ROWS, 512, 256, 0);                      // q|k proj
    gemm(tgt, w_in_a + 512 * 256, b_in_a + 512, SC, ROWS, 256, 256, 0);   // v proj
    attn_intra_kernel<<<400 * 8, 128, 0, stream>>>(SB, SC, SD);
    gemm(SD, w_out_a, b_out_a, SE, ROWS, 256, 256, 0);
    residual_ln_kernel<<<LN_BLOCKS, 256, 0, stream>>>(SE, tgt, g_ln_a, b_ln_a, SF, ROWS);

    // ---- inter-instance self-attention over queries ----
    transpose_qp_kernel<<<EW_BLOCKS, 256, 0, stream>>>(SF, SG, 1);        // ti (b,p,q)
    gemm(SG, w_in_i, b_in_i, SH, ROWS, 768, 256, 0);                      // qkv proj
    attn_inter_kernel<<<100 * 8, 256, 0, stream>>>(SH, SD);
    gemm(SD, w_out_i, b_out_i, SE, ROWS, 256, 256, 0);
    residual_ln_kernel<<<LN_BLOCKS, 256, 0, stream>>>(SE, SG, g_ln_i, b_ln_i, SF, ROWS);
    transpose_qp_kernel<<<EW_BLOCKS, 256, 0, stream>>>(SF, SG, 0);        // back to (b,q,p)

    // ---- deformable cross-attention ----
    add_kernel<<<EW_BLOCKS, 256, 0, stream>>>(SG, qpos, SA, NEL);         // qc
    gemm(SA, off_w, off_b, SJ, ROWS, 256, 256, 0);                        // offsets
    gemm(SA, aw_w, aw_b, SI, ROWS, 128, 256, 0);                          // attn weights
    aw_softmax_kernel<<<(ROWS * NH + 255) / 256, 256, 0, stream>>>(SI);
    deform_sample_kernel<<<SAMP_BLOCKS, 256, 0, stream>>>(VAL, SJ, SI, refp, SD);
    gemm(SD, co_w, co_b, SE, ROWS, 256, 256, 0);                          // output proj
    residual_ln_kernel<<<LN_BLOCKS, 256, 0, stream>>>(SE, SG, g_ln_c, b_ln_c, SK, ROWS);

    // ---- FFN ----
    gemm(SK, lin1_w, lin1_b, SH, ROWS, 1024, 256, 1);                     // relu
    gemm(SH, lin2_w, lin2_b, SE, ROWS, 256, 1024, 0);
    residual_ln_kernel<<<LN_BLOCKS, 256, 0, stream>>>(SE, SK, g_ln3, b_ln3,
                                                      (float*)d_out, ROWS);
}
